// CausalSelfAttention_64338610094602
// MI455X (gfx1250) — compile-verified, matrix operand order checked by reading, not run on hardware
//
#include <hip/hip_runtime.h>

// ---------------------------------------------------------------------------
// Problem constants (from reference)
// ---------------------------------------------------------------------------
#define B_  4
#define T_  2048
#define C_  2048
#define H_  16
#define D_  128
#define BT_ (B_ * T_)

typedef __attribute__((ext_vector_type(16))) __bf16 v16bf;
typedef __attribute__((ext_vector_type(8)))  __bf16 v8bf;
typedef __attribute__((ext_vector_type(8)))  float  v8f;

// ---------------------------------------------------------------------------
// CDNA5 async global->LDS path (ASYNCcnt).  Guarded so the file compiles on
// toolchains without the builtin (falls back to load + ds_store).
// Builtin signature (from clang diagnostic): param 0 is v4i32 in AS1 (global),
// param 1 is the LDS destination (AS3), then imm offset + imm cpol.
// ---------------------------------------------------------------------------
#if defined(__has_builtin)
#if __has_builtin(__builtin_amdgcn_global_load_async_to_lds_b128)
#define HAS_ASYNC_LDS 1
#endif
#endif

typedef int v4i_vec __attribute__((vector_size(16)));
typedef __attribute__((address_space(1))) v4i_vec as1_v4i;
typedef __attribute__((address_space(3))) v4i_vec as3_v4i;
#define GPTR(p) ((as1_v4i*)(p))
#define LPTR(p) ((as3_v4i*)(p))

// ---------------------------------------------------------------------------
// bf16 helpers (round-to-nearest-even)
// ---------------------------------------------------------------------------
__device__ __forceinline__ __bf16 f2bf(float f) {
  unsigned u = __builtin_bit_cast(unsigned, f);
  unsigned r = u + 0x7FFFu + ((u >> 16) & 1u);
  unsigned short s = (unsigned short)(r >> 16);
  return __builtin_bit_cast(__bf16, s);
}
__device__ __forceinline__ float bf2f(__bf16 b) {
  unsigned short s = __builtin_bit_cast(unsigned short, b);
  unsigned u = ((unsigned)s) << 16;
  return __builtin_bit_cast(float, u);
}

// ---------------------------------------------------------------------------
// WMMA 16x16x32 bf16 fragment loader.
// 16-bit A/B layout (ISA 7.12.2): lanes 0-15 hold K=[k0..k0+7] (VGPR0-3) and
// K=[k0+16..k0+23] (VGPR4-7); lanes 16-31 hold K=[k0+8..k0+15], [k0+24..k0+31].
// Two 16B loads per lane. `row` must already include (lane & 15) selection.
// ---------------------------------------------------------------------------
__device__ __forceinline__ v16bf load_frag(const __bf16* base, int ld, int row, int k0) {
  int lane = threadIdx.x & 31;
  int kb = k0 + ((lane & 16) ? 8 : 0);
  const __bf16* p = base + (size_t)row * ld + kb;
  v8bf lo = *(const v8bf*)(p);
  v8bf hi = *(const v8bf*)(p + 16);
  v16bf r;
#pragma unroll
  for (int i = 0; i < 8; i++) { r[i] = lo[i]; r[i + 8] = hi[i]; }
  return r;
}

__device__ __forceinline__ v8f wmma_bf16(v16bf a, v16bf b, v8f c) {
  return __builtin_amdgcn_wmma_f32_16x16x32_bf16(false, a, false, b, (short)0, c,
                                                 false, false);
}

// ---------------------------------------------------------------------------
// Async staging helpers: 64B row copy and 16B chunk copy, global -> LDS.
// ---------------------------------------------------------------------------
__device__ __forceinline__ void stage_row64(const __bf16* g, __bf16* l) {
#if HAS_ASYNC_LDS
  __builtin_amdgcn_global_load_async_to_lds_b128(GPTR(g), LPTR(l), 0, 0);
  __builtin_amdgcn_global_load_async_to_lds_b128(GPTR(g), LPTR(l), 16, 0);
  __builtin_amdgcn_global_load_async_to_lds_b128(GPTR(g), LPTR(l), 32, 0);
  __builtin_amdgcn_global_load_async_to_lds_b128(GPTR(g), LPTR(l), 48, 0);
#else
#pragma unroll
  for (int c = 0; c < 4; c++) *(v8bf*)(l + c * 8) = *(const v8bf*)(g + c * 8);
#endif
}
__device__ __forceinline__ void stage_chunk16(const __bf16* g, __bf16* l) {
#if HAS_ASYNC_LDS
  __builtin_amdgcn_global_load_async_to_lds_b128(GPTR(g), LPTR(l), 0, 0);
#else
  *(v8bf*)l = *(const v8bf*)g;
#endif
}
__device__ __forceinline__ void async_wait_all() {
#if HAS_ASYNC_LDS
  asm volatile("s_wait_asynccnt 0x0" ::: "memory");
#endif
}

// ---------------------------------------------------------------------------
// fp32 -> bf16 conversion (grid-strided)
// ---------------------------------------------------------------------------
__global__ void cvt_f32_bf16(const float* __restrict__ in, __bf16* __restrict__ out,
                             long n) {
  long i = (long)blockIdx.x * blockDim.x + threadIdx.x;
  long stride = (long)gridDim.x * blockDim.x;
  for (; i < n; i += stride) out[i] = f2bf(in[i]);
}

// ---------------------------------------------------------------------------
// omega[b,t] = sigmoid((x[b,t,:] . w_omega + b_omega) / 16)   (one wave / row)
// ---------------------------------------------------------------------------
__global__ void omega_kernel(const float* __restrict__ x, const float* __restrict__ w,
                             const float* __restrict__ bias, float* __restrict__ omega,
                             int rows) {
  int wid = (int)((blockIdx.x * blockDim.x + threadIdx.x) >> 5);
  int lane = threadIdx.x & 31;
  if (wid >= rows) return;
  const float* xr = x + (size_t)wid * C_;
  float s = 0.f;
  for (int i = lane; i < C_; i += 32) s += xr[i] * w[i];
#pragma unroll
  for (int off = 16; off; off >>= 1) s += __shfl_xor(s, off, 32);
  if (lane == 0) {
    float z = (s + bias[0]) * (1.0f / 16.0f);
    omega[wid] = 1.0f / (1.0f + __expf(-z));
  }
}

// ---------------------------------------------------------------------------
// phi = exclusive cumsum of omega along T.  One 256-thread block per batch.
// ---------------------------------------------------------------------------
__global__ void scan_kernel(const float* __restrict__ omega, float* __restrict__ phi) {
  __shared__ float sums[256];
  int b = blockIdx.x, t = threadIdx.x;
  const float* om = omega + (size_t)b * T_;
  float* ph = phi + (size_t)b * T_;
  float v[8];
  float run = 0.f;
  int base = t * 8;
#pragma unroll
  for (int i = 0; i < 8; i++) { v[i] = om[base + i]; run += v[i]; }
  sums[t] = run;
  __syncthreads();
  for (int off = 1; off < 256; off <<= 1) {
    float y = (t >= off) ? sums[t - off] : 0.f;
    __syncthreads();
    sums[t] += y;
    __syncthreads();
  }
  float excl = sums[t] - run;  // exclusive prefix of this thread's chunk
  float acc = excl;
#pragma unroll
  for (int i = 0; i < 8; i++) { ph[base + i] = acc; acc += v[i]; }
}

// ---------------------------------------------------------------------------
// WMMA GEMM:  Out[M,N] = A[M,K] (bf16) x W[N,K]^T (bf16)
// MODE 0: bf16 row-major out.  MODE 1: bf16 out transposed to [b][h][d][t] (V).
// MODE 2: fp32 row-major out (final projection).
// Block = 256 threads = 8 waves; block tile 256 x 64; K-step 32.
// A (256x32 = 16KB) and B (64x32 = 4KB) tiles staged in LDS via CDNA5 async
// global->LDS loads (ASYNCcnt), fragments re-read with ds_load_b128.
// Next K-tile prefetched (global_prefetch_b8) while WMMAs run.
// ---------------------------------------------------------------------------
template <int MODE>
__launch_bounds__(256)
__global__ void gemm_wmma(const __bf16* __restrict__ A, const __bf16* __restrict__ W,
                          void* __restrict__ Out, int M, int N, int K) {
  __shared__ __align__(16) __bf16 Asmem[256 * 32];
  __shared__ __align__(16) __bf16 Bsmem[64 * 32];
  int t = threadIdx.x;
  int w = t >> 5;
  int lane = t & 31;
  int ln = lane & 15, half = lane >> 4;
  int mblk = blockIdx.x * 256;
  int nblk = blockIdx.y * 64;
  int brow = t >> 2, bcol = (t & 3) * 8;

  v8f acc[2][4] = {};
  for (int k0 = 0; k0 < K; k0 += 32) {
    // stage tiles into LDS (async on CDNA5)
    stage_row64(A + (size_t)(mblk + t) * K + k0, &Asmem[t * 32]);
    stage_chunk16(W + (size_t)(nblk + brow) * K + k0 + bcol,
                  &Bsmem[brow * 32 + bcol]);
    if (k0 + 32 < K) {
      __builtin_prefetch(A + (size_t)(mblk + t) * K + k0 + 32, 0, 0);
      __builtin_prefetch(W + (size_t)(nblk + brow) * K + k0 + 32 + bcol, 0, 0);
    }
    async_wait_all();
    __syncthreads();

    v16bf a0 = load_frag(Asmem, 32, w * 32 + ln, 0);
    v16bf a1 = load_frag(Asmem, 32, w * 32 + 16 + ln, 0);
    v16bf bfrag[4];
#pragma unroll
    for (int j = 0; j < 4; j++) bfrag[j] = load_frag(Bsmem, 32, 16 * j + ln, 0);
#pragma unroll
    for (int j = 0; j < 4; j++) {
      acc[0][j] = wmma_bf16(a0, bfrag[j], acc[0][j]);
      acc[1][j] = wmma_bf16(a1, bfrag[j], acc[1][j]);
    }
    __syncthreads();
  }
  // C/D layout: VGPR i -> row (i | i+8 by lane half), col = n0 + (lane&15)
#pragma unroll
  for (int mi = 0; mi < 2; mi++)
#pragma unroll
    for (int j = 0; j < 4; j++)
#pragma unroll
      for (int i = 0; i < 8; i++) {
        int row = mblk + w * 32 + 16 * mi + i + half * 8;
        int col = nblk + 16 * j + ln;
        float vv = acc[mi][j][i];
        if (MODE == 0) {
          ((__bf16*)Out)[(size_t)row * N + col] = f2bf(vv);
        } else if (MODE == 2) {
          ((float*)Out)[(size_t)row * N + col] = vv;
        } else {  // V transposed: [b][h][d][t]
          int bb = row / T_, tt = row % T_;
          int hh = col >> 7, dd = col & 127;
          ((__bf16*)Out)[(((size_t)bb * H_ + hh) * D_ + dd) * T_ + tt] = f2bf(vv);
        }
      }
}

// ---------------------------------------------------------------------------
// Gated rotary + RMSNorm, in place on bf16 Q or K.  One wave per (b,t,h) row.
// Lane handles pairs (d, d+64) for d = lane and lane+32.
// ---------------------------------------------------------------------------
__global__ void rope_rms_kernel(__bf16* __restrict__ Q, const float* __restrict__ phi,
                                const float* __restrict__ log_freq,
                                const float* __restrict__ gamma) {
  int gwid = (int)((blockIdx.x * blockDim.x + threadIdx.x) >> 5);
  int lane = threadIdx.x & 31;
  if (gwid >= BT_ * H_) return;
  int bt = gwid / H_, h = gwid % H_;
  __bf16* row = Q + ((size_t)bt * H_ + h) * D_;
  float p = phi[bt];
  float vals[4];
  float sumsq = 0.f;
#pragma unroll
  for (int j = 0; j < 2; j++) {
    int d = lane + 32 * j;
    float freq = __expf(log_freq[d]);
    float ang = p * freq;
    float s, c;
    __sincosf(ang, &s, &c);
    float t1 = bf2f(row[d]);
    float t2 = bf2f(row[d + 64]);
    float n1 = t1 * c + t2 * s;
    float n2 = -t1 * s + t2 * c;
    vals[2 * j] = n1;
    vals[2 * j + 1] = n2;
    sumsq += n1 * n1 + n2 * n2;
  }
#pragma unroll
  for (int off = 16; off; off >>= 1) sumsq += __shfl_xor(sumsq, off, 32);
  float scale = rsqrtf(sumsq * (1.0f / (float)D_) + 1e-5f);
#pragma unroll
  for (int j = 0; j < 2; j++) {
    int d = lane + 32 * j;
    row[d] = f2bf(vals[2 * j] * scale * gamma[d]);
    row[d + 64] = f2bf(vals[2 * j + 1] * scale * gamma[d + 64]);
  }
}

// ---------------------------------------------------------------------------
// Flash attention (causal, online softmax).  Block = 8 waves; wave owns a
// 16-query tile; key tiles of 32.  Scores: 8 x wmma (K=128).  P transposed
// through per-wave LDS (A-layout reload), P.V: 8 x wmma using transposed V.
// ---------------------------------------------------------------------------
__launch_bounds__(256)
__global__ void attn_kernel(const __bf16* __restrict__ Q, const __bf16* __restrict__ Km,
                            const __bf16* __restrict__ Vt, __bf16* __restrict__ Y) {
  __shared__ __align__(16) __bf16 Pl[8][16 * 32];
  int w = threadIdx.x >> 5;
  int lane = threadIdx.x & 31;
  int ln = lane & 15, half = lane >> 4;
  int q0 = blockIdx.x * 128 + w * 16;
  int h = blockIdx.y, b = blockIdx.z;
  const int ldq = H_ * D_;
  const __bf16* Qb = Q + ((size_t)b * T_ * H_ + h) * D_;
  const __bf16* Kb = Km + ((size_t)b * T_ * H_ + h) * D_;
  const __bf16* Vb = Vt + ((size_t)b * H_ + h) * (size_t)D_ * T_;

  v16bf qf[4];
#pragma unroll
  for (int kk = 0; kk < 4; kk++) qf[kk] = load_frag(Qb, ldq, q0 + ln, kk * 32);

  v8f yacc[8] = {};
  float mrow[8], lrow[8];
#pragma unroll
  for (int i = 0; i < 8; i++) { mrow[i] = -3.0e38f; lrow[i] = 0.f; }

  const float scl = 0.08838834764831845f;  // 1/sqrt(128)
  int ntiles = (q0 + 16 + 31) >> 5;
  for (int kt = 0; kt < ntiles; kt++) {
    int s0 = kt * 32;
    v8f sc[2] = {};
#pragma unroll
    for (int kk = 0; kk < 4; kk++) {
      v16bf b0 = load_frag(Kb, ldq, s0 + ln, kk * 32);
      v16bf b1 = load_frag(Kb, ldq, s0 + 16 + ln, kk * 32);
      sc[0] = wmma_bf16(qf[kk], b0, sc[0]);
      sc[1] = wmma_bf16(qf[kk], b1, sc[1]);
    }
    // scale + causal mask (C layout: VGPR i -> row i|i+8, col = s0+16j+ln)
#pragma unroll
    for (int j = 0; j < 2; j++) {
      int scol = s0 + 16 * j + ln;
#pragma unroll
      for (int i = 0; i < 8; i++) {
        int trow = q0 + i + half * 8;
        float vv = sc[j][i] * scl;
        if (scol > trow) vv = -3.0e38f;
        sc[j][i] = vv;
      }
    }
    // online softmax: row max / exp / row sum (reduce across 16-lane halves)
#pragma unroll
    for (int i = 0; i < 8; i++) {
      float m_ = fmaxf(sc[0][i], sc[1][i]);
#pragma unroll
      for (int off = 8; off; off >>= 1) m_ = fmaxf(m_, __shfl_xor(m_, off, 32));
      float newm = fmaxf(mrow[i], m_);
      float corr = __expf(mrow[i] - newm);
      float e0 = __expf(sc[0][i] - newm);
      float e1 = __expf(sc[1][i] - newm);
      sc[0][i] = e0;
      sc[1][i] = e1;
      float rs = e0 + e1;
#pragma unroll
      for (int off = 8; off; off >>= 1) rs += __shfl_xor(rs, off, 32);
      lrow[i] = lrow[i] * corr + rs;
      mrow[i] = newm;
#pragma unroll
      for (int j = 0; j < 8; j++) yacc[j][i] *= corr;
    }
    // transpose P through LDS into A-matrix layout
#pragma unroll
    for (int j = 0; j < 2; j++)
#pragma unroll
      for (int i = 0; i < 8; i++)
        Pl[w][(i + half * 8) * 32 + 16 * j + ln] = f2bf(sc[j][i]);
    asm volatile("s_wait_dscnt 0" ::: "memory");
    v16bf pf = load_frag(&Pl[w][0], 32, ln, 0);
    // P (16x32) x V (32x128): Vt rows are d, K=s contiguous
#pragma unroll
    for (int dj = 0; dj < 8; dj++) {
      v16bf vf = load_frag(Vb, T_, dj * 16 + ln, s0);
      yacc[dj] = wmma_bf16(pf, vf, yacc[dj]);
    }
    asm volatile("s_wait_dscnt 0" ::: "memory");
  }
  // finalize and store Y[b,t,h*D+d] (bf16, feeds output GEMM)
  __bf16* Yb = Y + ((size_t)b * T_ * H_ + h) * D_;
#pragma unroll
  for (int dj = 0; dj < 8; dj++)
#pragma unroll
    for (int i = 0; i < 8; i++) {
      int trow = q0 + i + half * 8;
      int col = dj * 16 + ln;
      Yb[(size_t)trow * ldq + col] = f2bf(yacc[dj][i] / lrow[i]);
    }
}

// ---------------------------------------------------------------------------
// Host launcher
// ---------------------------------------------------------------------------
extern "C" void kernel_launch(void* const* d_in, const int* in_sizes, int n_in,
                              void* d_out, int out_size, void* d_ws, size_t ws_size,
                              hipStream_t stream) {
  const float* x       = (const float*)d_in[0];
  const float* Wq      = (const float*)d_in[1];
  const float* Wk      = (const float*)d_in[2];
  const float* Wv      = (const float*)d_in[3];
  const float* Wo      = (const float*)d_in[4];
  const float* w_omega = (const float*)d_in[5];
  const float* b_omega = (const float*)d_in[6];
  const float* logfreq = (const float*)d_in[7];
  const float* q_gamma = (const float*)d_in[8];
  const float* k_gamma = (const float*)d_in[9];
  float* out = (float*)d_out;

  // workspace carve-up
  char* ws = (char*)d_ws;
  size_t off = 0;
  auto carve = [&](size_t bytes) {
    void* p = ws + off;
    off += (bytes + 255) & ~(size_t)255;
    return p;
  };
  float*  omega = (float*)carve((size_t)BT_ * 4);
  float*  phi   = (float*)carve((size_t)BT_ * 4);
  __bf16* xb    = (__bf16*)carve((size_t)BT_ * C_ * 2);
  __bf16* wqb   = (__bf16*)carve((size_t)C_ * C_ * 2);
  __bf16* wkb   = (__bf16*)carve((size_t)C_ * C_ * 2);
  __bf16* wvb   = (__bf16*)carve((size_t)C_ * C_ * 2);
  __bf16* wob   = (__bf16*)carve((size_t)C_ * C_ * 2);
  __bf16* qb    = (__bf16*)carve((size_t)BT_ * C_ * 2);
  __bf16* kb    = (__bf16*)carve((size_t)BT_ * C_ * 2);
  __bf16* vt    = (__bf16*)carve((size_t)BT_ * C_ * 2);
  __bf16* yb    = (__bf16*)carve((size_t)BT_ * C_ * 2);

  // 1) bf16 conversions
  cvt_f32_bf16<<<8192, 256, 0, stream>>>(x, xb, (long)BT_ * C_);
  cvt_f32_bf16<<<4096, 256, 0, stream>>>(Wq, wqb, (long)C_ * C_);
  cvt_f32_bf16<<<4096, 256, 0, stream>>>(Wk, wkb, (long)C_ * C_);
  cvt_f32_bf16<<<4096, 256, 0, stream>>>(Wv, wvb, (long)C_ * C_);
  cvt_f32_bf16<<<4096, 256, 0, stream>>>(Wo, wob, (long)C_ * C_);

  // 2) gate + exclusive phase scan
  omega_kernel<<<BT_ / 8, 256, 0, stream>>>(x, w_omega, b_omega, omega, BT_);
  scan_kernel<<<B_, 256, 0, stream>>>(omega, phi);

  // 3) QKV projections (WMMA bf16); V stored transposed [b][h][d][t]
  dim3 ggrid(BT_ / 256, C_ / 64);
  gemm_wmma<0><<<ggrid, 256, 0, stream>>>(xb, wqb, qb, BT_, C_, C_);
  gemm_wmma<0><<<ggrid, 256, 0, stream>>>(xb, wkb, kb, BT_, C_, C_);
  gemm_wmma<1><<<ggrid, 256, 0, stream>>>(xb, wvb, vt, BT_, C_, C_);

  // 4) gated rotary + RMSNorm, in place
  rope_rms_kernel<<<BT_ * H_ / 8, 256, 0, stream>>>(qb, phi, logfreq, q_gamma);
  rope_rms_kernel<<<BT_ * H_ / 8, 256, 0, stream>>>(kb, phi, logfreq, k_gamma);

  // 5) causal flash attention
  attn_kernel<<<dim3(T_ / 128, H_, B_), 256, 0, stream>>>(qb, kb, vt, yb);

  // 6) output projection (fp32 result)
  gemm_wmma<2><<<ggrid, 256, 0, stream>>>(yb, wob, out, BT_, C_, C_);
}
